// CrossAttention_21509196218969
// MI455X (gfx1250) — matrix operand order repaired and checked by reading, not compiled, not run on hardware
//
#include <hip/hip_runtime.h>

typedef unsigned short u16;
typedef __bf16 bf16_t;
typedef __attribute__((ext_vector_type(16))) bf16_t v16bf;
typedef __attribute__((ext_vector_type(8)))  float  v8f;
typedef __attribute__((ext_vector_type(4)))  unsigned int u32x4;
typedef __attribute__((ext_vector_type(8)))  int i32x8;
typedef __attribute__((ext_vector_type(4)))  int i32x4;

union Frag {
    uint4 u[2];
    v16bf v;
    u16   h[16];
};

__device__ __forceinline__ u16 f2bf(float f) {
    unsigned int u = __float_as_uint(f);
    unsigned int r = u + 0x7FFFu + ((u >> 16) & 1u);
    return (u16)(r >> 16);
}

__device__ __forceinline__ v8f v8f_zero() {
    v8f z = {0.f, 0.f, 0.f, 0.f, 0.f, 0.f, 0.f, 0.f};
    return z;
}

// ---------------------------------------------------------------------------
// f32 -> bf16 pack
// ---------------------------------------------------------------------------
__global__ void pack_bf16_kernel(const float* __restrict__ src, u16* __restrict__ dst, int n) {
    int i = blockIdx.x * blockDim.x + threadIdx.x;
    if (i < n) dst[i] = f2bf(src[i]);
}

// ---------------------------------------------------------------------------
// GEMM: Y[m,n] = sum_k A[m,k] * W[n,k] + bias[n]   (K = N = 768 fixed)
// A: bf16 M x 768 row-major.  W: bf16 768 x 768 row-major (original layout).
// OBF ? output bf16 : output f32.
// Block: 256 threads (8 waves), 128x128 tile, K staged in 32-chunks via LDS.
// ---------------------------------------------------------------------------
template <bool OBF>
__global__ __launch_bounds__(256) void gemm_bf16_768(
    const u16* __restrict__ A, const u16* __restrict__ W,
    const float* __restrict__ bias, void* __restrict__ Yv, int M) {
    __shared__ __align__(16) u16 As[128][40];  // row stride 80B (16B multiple)
    __shared__ __align__(16) u16 Bs[128][40];

    const int tid  = threadIdx.x;
    const int lane = tid & 31;
    const int wave = tid >> 5;
    const int wm   = wave & 1;   // 64-row half
    const int wn   = wave >> 1;  // 32-col slice
    const int m0   = blockIdx.y * 128;
    const int n0   = blockIdx.x * 128;
    const int lr   = lane & 15;
    const int asel = (lane < 16) ? 0 : 8;    // A-frag K offset (elems)
    const int bsel = (lane < 16) ? 0 : 8;    // B-frag K offset (elems)

    v8f acc[4][2];
#pragma unroll
    for (int i = 0; i < 4; i++)
#pragma unroll
        for (int j = 0; j < 2; j++) acc[i][j] = v8f_zero();

    for (int k0 = 0; k0 < 768; k0 += 32) {
#pragma unroll
        for (int t = 0; t < 2; t++) {
            int i = tid + t * 256;      // 512 uint4 per tile
            int row = i >> 2, seg = i & 3;
            uint4 va = make_uint4(0u, 0u, 0u, 0u);
            if (m0 + row < M) {
                const u16* p = A + (size_t)(m0 + row) * 768 + k0 + seg * 8;
                va = *reinterpret_cast<const uint4*>(p);
                if (k0 + 32 < 768)
                    __builtin_prefetch((const void*)(p + 32), 0, 1);  // global_prefetch_b8
            }
            *reinterpret_cast<uint4*>(&As[row][seg * 8]) = va;
            uint4 vb = *reinterpret_cast<const uint4*>(W + (size_t)(n0 + row) * 768 + k0 + seg * 8);
            *reinterpret_cast<uint4*>(&Bs[row][seg * 8]) = vb;
        }
        __syncthreads();

        Frag a[4];
#pragma unroll
        for (int mt = 0; mt < 4; mt++) {
            int r = wm * 64 + mt * 16 + lr;
            a[mt].u[0] = *reinterpret_cast<const uint4*>(&As[r][asel]);        // K 0-7 / 8-15
            a[mt].u[1] = *reinterpret_cast<const uint4*>(&As[r][16 + asel]);   // K 16-23 / 24-31
        }
#pragma unroll
        for (int nt = 0; nt < 2; nt++) {
            Frag bf;
            int n = wn * 32 + nt * 16 + lr;
            bf.u[0] = *reinterpret_cast<const uint4*>(&Bs[n][bsel]);       // K 0-7 / 8-15
            bf.u[1] = *reinterpret_cast<const uint4*>(&Bs[n][bsel + 16]);  // K 16-23 / 24-31
#pragma unroll
            for (int mt = 0; mt < 4; mt++)
                acc[mt][nt] = __builtin_amdgcn_wmma_f32_16x16x32_bf16(
                    false, a[mt].v, false, bf.v, (short)0, acc[mt][nt], false, false);
        }
        __syncthreads();
    }

#pragma unroll
    for (int mt = 0; mt < 4; mt++)
#pragma unroll
        for (int nt = 0; nt < 2; nt++)
#pragma unroll
            for (int r = 0; r < 8; r++) {
                int row = m0 + wm * 64 + mt * 16 + r + ((lane & 16) ? 8 : 0);
                int col = n0 + wn * 32 + nt * 16 + lr;
                if (row < M) {
                    float v = acc[mt][nt][r] + bias[col];
                    if (OBF)
                        ((u16*)Yv)[(size_t)row * 768 + col] = f2bf(v);
                    else
                        ((float*)Yv)[(size_t)row * 768 + col] = v;
                }
            }
}

// ---------------------------------------------------------------------------
// Head-split gathers (faithful to the reference's transpose->reshape quirk).
// q_heads[b,h,t',d] = Q[b, flat % Tq, flat / Tq], flat = t'*768 + h*64 + d
// ---------------------------------------------------------------------------
__global__ void gather_q_kernel(const u16* __restrict__ Qf, u16* __restrict__ qg) {
    int idx = blockIdx.x * blockDim.x + threadIdx.x;  // B*H*4096*64 = 25165824
    int d  = idx & 63;
    int r1 = idx >> 6;
    int tp = r1 & 4095;
    int bh = r1 >> 12;
    int h  = bh % 12;
    int b  = bh / 12;
    int flat = tp * 768 + h * 64 + d;
    int t = flat & 4095;
    int c = flat >> 12;
    qg[idx] = Qf[((size_t)(b << 12) + t) * 768 + c];
}

__global__ void gather_kv_kernel(const u16* __restrict__ Kf, const u16* __restrict__ Vf,
                                 u16* __restrict__ kg, u16* __restrict__ vg) {
    int idx = blockIdx.x * blockDim.x + threadIdx.x;  // B*H*77*64 = 473088
    int d  = idx & 63;
    int r1 = idx >> 6;
    int kk = r1 % 77;
    int bh = r1 / 77;
    int h  = bh % 12;
    int b  = bh / 12;
    int flat = kk * 768 + h * 64 + d;
    int t = flat % 77;
    int c = flat / 77;
    size_t src = ((size_t)b * 77 + t) * 768 + c;
    kg[idx] = Kf[src];
    vg[idx] = Vf[src];
}

// ---------------------------------------------------------------------------
// Attention: one wave per 16 query rows; Tkv = 77 padded to 80 (S) / 96 (K-dim).
// grid = (32, B*H); block = 256.
// K tile is staged into LDS by the Tensor Data Mover when available:
//   - 1-D run of 80*64 bf16 elements, tensor_dim0 = 77*64 so the hardware
//     zero-fills pad rows 77..79 (OOB reads return zero),
//   - pad_enable inserts 4 DWORDs (16B) every 32 DWORDs (128B) so the LDS
//     image lands directly in the padded Kl[80][72] layout.
// ---------------------------------------------------------------------------
__global__ __launch_bounds__(256) void attn_kernel(
    const u16* __restrict__ qg, const u16* __restrict__ kg,
    const u16* __restrict__ vg, u16* __restrict__ om) {
    __shared__ __align__(16) u16 Kl[80][72];       // [k_idx][d]
    __shared__ __align__(16) u16 Vl[64][104];      // [d][k_idx] (cols 0..95 valid)
    __shared__ __align__(16) u16 Pl[8][16][104];   // per-wave probs, cols 0..95

    const int tid  = threadIdx.x;
    const int lane = tid & 31;
    const int w    = tid >> 5;
    const int bh   = blockIdx.y;  // b*12 + h
    const int b    = bh / 12;
    const int h    = bh % 12;
    const int q0   = blockIdx.x * 128 + w * 16;
    const int lr   = lane & 15;
    const int hi   = (lane & 16) ? 1 : 0;

#if defined(__gfx1250__) && __has_builtin(__builtin_amdgcn_tensor_load_to_lds)
    if (w == 0) {
        // D# group 0: count=1 (valid), lds_addr, global_addr, type=2 ("image")
        unsigned long long ga = (unsigned long long)(uintptr_t)(kg + (size_t)bh * 77 * 64);
        unsigned lds = (unsigned)(uintptr_t)&Kl[0][0];  // flat LDS ptr low 32 bits = LDS offset
        u32x4 g0 = {1u, lds,
                    (unsigned)(ga & 0xFFFFFFFFu),
                    (unsigned)((ga >> 32) & 0x01FFFFFFu) | (2u << 30)};
        // D# group 1: data_size=1 (2B), pad_enable, pad_interval=4 (32 DW = 128B),
        // pad_amount=3 (4 DW = 16B); tensor_dim0=4928, tensor_dim1=1,
        // tile_dim0=5120, tile_dim1=1.
        i32x8 g1 = {(int)((1u << 16) | (1u << 20) | (4u << 22) | (3u << 25)),
                    (int)(4928u << 16),                 // dim0[15:0] << 16 (bar addr = 0)
                    (int)((4928u >> 16) | (1u << 16)),  // dim0[31:16] | dim1[15:0] << 16
                    (int)(5120u << 16),                 // dim1[31:16]=0 | tile_dim0 << 16
                    1,                                  // tile_dim1=1, tile_dim2=0
                    0, 0, 0};                           // strides unused (single row)
        i32x4 gz4 = {0, 0, 0, 0};
        i32x8 gz8 = {0, 0, 0, 0, 0, 0, 0, 0};
        __builtin_amdgcn_tensor_load_to_lds(g0, g1, gz4, gz4, gz8, 0);
        __builtin_amdgcn_s_wait_tensorcnt(0);
    }
#else
    for (int i = tid; i < 80 * 64; i += 256) {
        int kk = i >> 6, d = i & 63;
        u16 v = 0;
        if (kk < 77) v = kg[((size_t)bh * 77 + kk) * 64 + d];
        Kl[kk][d] = v;
    }
#endif
    for (int i = tid; i < 96 * 64; i += 256) {
        int kk = i >> 6, d = i & 63;
        u16 v = 0;
        if (kk < 77) v = vg[((size_t)bh * 77 + kk) * 64 + d];
        Vl[d][kk] = v;
    }
    __syncthreads();

    // Q A-fragments straight from global (rows are contiguous bf16x64)
    Frag aq[2];
    const u16* qrow = qg + ((size_t)bh * 4096 + q0 + lr) * 64;
#pragma unroll
    for (int kc = 0; kc < 2; kc++) {
        aq[kc].u[0] = *reinterpret_cast<const uint4*>(qrow + kc * 32 + (hi ? 8 : 0));
        aq[kc].u[1] = *reinterpret_cast<const uint4*>(qrow + kc * 32 + 16 + (hi ? 8 : 0));
    }

    // S = Q * K^T  (16 x 80)
    v8f s[5];
#pragma unroll
    for (int nt = 0; nt < 5; nt++) s[nt] = v8f_zero();
#pragma unroll
    for (int nt = 0; nt < 5; nt++) {
        int n = nt * 16 + lr;
#pragma unroll
        for (int kc = 0; kc < 2; kc++) {
            Frag kb;
            kb.u[0] = *reinterpret_cast<const uint4*>(&Kl[n][kc * 32 + (hi ? 8 : 0)]);
            kb.u[1] = *reinterpret_cast<const uint4*>(&Kl[n][kc * 32 + 16 + (hi ? 8 : 0)]);
            s[nt] = __builtin_amdgcn_wmma_f32_16x16x32_bf16(
                false, aq[kc].v, false, kb.v, (short)0, s[nt], false, false);
        }
    }

    const float scale = 0.03608439182435161f;  // 1/sqrt(768)  (ref uses sqrt(d_prompt))
#pragma unroll
    for (int nt = 0; nt < 5; nt++) {
        bool mask = (nt * 16 + lr) >= 77;
#pragma unroll
        for (int r = 0; r < 8; r++) {
            float v = s[nt][r] * scale;
            s[nt][r] = mask ? -1e30f : v;
        }
    }

    // row softmax (each half-wave owns 8 rows; width-16 butterfly reductions)
#pragma unroll
    for (int r = 0; r < 8; r++) {
        float mx = s[0][r];
#pragma unroll
        for (int nt = 1; nt < 5; nt++) mx = fmaxf(mx, s[nt][r]);
#pragma unroll
        for (int off = 8; off >= 1; off >>= 1) mx = fmaxf(mx, __shfl_xor(mx, off, 16));
        float sum = 0.f;
#pragma unroll
        for (int nt = 0; nt < 5; nt++) {
            float e = __expf(s[nt][r] - mx);
            s[nt][r] = e;
            sum += e;
        }
#pragma unroll
        for (int off = 8; off >= 1; off >>= 1) sum += __shfl_xor(sum, off, 16);
        float inv = 1.0f / sum;
#pragma unroll
        for (int nt = 0; nt < 5; nt++) s[nt][r] *= inv;
    }

    // C-layout -> A-layout via LDS (zero-pad K to 96)
#pragma unroll
    for (int r = 0; r < 8; r++) {
        int m = r + hi * 8;
#pragma unroll
        for (int nt = 0; nt < 5; nt++) Pl[w][m][nt * 16 + lr] = f2bf(s[nt][r]);
        Pl[w][m][80 + lr] = 0;
    }
    __syncthreads();

    // O = P * V   (16 x 64)
    v8f o[4];
#pragma unroll
    for (int nt = 0; nt < 4; nt++) o[nt] = v8f_zero();
#pragma unroll
    for (int kc = 0; kc < 3; kc++) {
        Frag pa;
        pa.u[0] = *reinterpret_cast<const uint4*>(&Pl[w][lr][kc * 32 + (hi ? 8 : 0)]);
        pa.u[1] = *reinterpret_cast<const uint4*>(&Pl[w][lr][kc * 32 + 16 + (hi ? 8 : 0)]);
#pragma unroll
        for (int nt = 0; nt < 4; nt++) {
            Frag vb;
            int n = nt * 16 + lr;
            vb.u[0] = *reinterpret_cast<const uint4*>(&Vl[n][kc * 32 + (hi ? 8 : 0)]);
            vb.u[1] = *reinterpret_cast<const uint4*>(&Vl[n][kc * 32 + 16 + (hi ? 8 : 0)]);
            o[nt] = __builtin_amdgcn_wmma_f32_16x16x32_bf16(
                false, pa.v, false, vb.v, (short)0, o[nt], false, false);
        }
    }

    // merged output layout (B, Tq, 768) as bf16 for the final projection
#pragma unroll
    for (int nt = 0; nt < 4; nt++)
#pragma unroll
        for (int r = 0; r < 8; r++) {
            int tq = q0 + r + hi * 8;
            int d  = nt * 16 + lr;
            om[((size_t)b * 4096 + tq) * 768 + h * 64 + d] = f2bf(o[nt][r]);
        }
}

// ---------------------------------------------------------------------------
// Host launch
// ---------------------------------------------------------------------------
extern "C" void kernel_launch(void* const* d_in, const int* in_sizes, int n_in,
                              void* d_out, int out_size, void* d_ws, size_t ws_size,
                              hipStream_t stream) {
    (void)in_sizes; (void)n_in; (void)out_size; (void)ws_size;
    const float* x      = (const float*)d_in[0];
    const float* prompt = (const float*)d_in[1];
    const float* Wq = (const float*)d_in[2];
    const float* bq = (const float*)d_in[3];
    const float* Wk = (const float*)d_in[4];
    const float* bk = (const float*)d_in[5];
    const float* Wv = (const float*)d_in[6];
    const float* bv = (const float*)d_in[7];
    const float* Wp = (const float*)d_in[8];
    const float* bp = (const float*)d_in[9];

    const int MQ  = 8 * 4096;        // 32768 query rows
    const int MKV = 8 * 77;          // 616 kv rows
    const size_t NQ  = (size_t)MQ * 768;   // 25165824
    const size_t NKV = (size_t)MKV * 768;  // 473088
    const size_t NW  = 768 * 768;          // 589824

    char* ws = (char*)d_ws;
    size_t off = 0;
    auto carve = [&](size_t bytes) {
        char* p = ws + off;
        off += (bytes + 255) & ~(size_t)255;
        return p;
    };
    u16* xb  = (u16*)carve(NQ * 2);   // x bf16; region reused as `om` after Q GEMM
    u16* pb  = (u16*)carve(NKV * 2);
    u16* wqb = (u16*)carve(NW * 2);
    u16* wkb = (u16*)carve(NW * 2);
    u16* wvb = (u16*)carve(NW * 2);
    u16* wpb = (u16*)carve(NW * 2);
    u16* Qf  = (u16*)carve(NQ * 2);
    u16* Kf  = (u16*)carve(NKV * 2);
    u16* Vf  = (u16*)carve(NKV * 2);
    u16* qg  = (u16*)carve(NQ * 2);
    u16* kg  = (u16*)carve(NKV * 2);
    u16* vg  = (u16*)carve(NKV * 2);
    u16* om  = xb;  // xb is dead after the Q projection

    // bf16 packs
    pack_bf16_kernel<<<(int)((NQ + 255) / 256), 256, 0, stream>>>(x, xb, (int)NQ);
    pack_bf16_kernel<<<(int)((NKV + 255) / 256), 256, 0, stream>>>(prompt, pb, (int)NKV);
    pack_bf16_kernel<<<(int)((NW + 255) / 256), 256, 0, stream>>>(Wq, wqb, (int)NW);
    pack_bf16_kernel<<<(int)((NW + 255) / 256), 256, 0, stream>>>(Wk, wkb, (int)NW);
    pack_bf16_kernel<<<(int)((NW + 255) / 256), 256, 0, stream>>>(Wv, wvb, (int)NW);
    pack_bf16_kernel<<<(int)((NW + 255) / 256), 256, 0, stream>>>(Wp, wpb, (int)NW);

    // projections (bf16 out)
    gemm_bf16_768<true><<<dim3(6, MQ / 128), 256, 0, stream>>>(xb, wqb, bq, Qf, MQ);
    gemm_bf16_768<true><<<dim3(6, (MKV + 127) / 128), 256, 0, stream>>>(pb, wkb, bk, Kf, MKV);
    gemm_bf16_768<true><<<dim3(6, (MKV + 127) / 128), 256, 0, stream>>>(pb, wvb, bv, Vf, MKV);

    // head-split gathers (reference layout quirk)
    gather_q_kernel<<<(int)(NQ / 256), 256, 0, stream>>>(Qf, qg);
    gather_kv_kernel<<<(int)((NKV + 255) / 256), 256, 0, stream>>>(Kf, Vf, kg, vg);

    // attention
    attn_kernel<<<dim3(32, 8 * 12), 256, 0, stream>>>(qg, kg, vg, om);

    // output projection (f32 out -> d_out)
    gemm_bf16_768<false><<<dim3(6, MQ / 128), 256, 0, stream>>>(om, wpb, bp, (float*)d_out, MQ);
}
